// ScreeningUnit_33724083208371
// MI455X (gfx1250) — matrix operand order repaired and checked by reading, not compiled, not run on hardware
//
#include <hip/hip_runtime.h>
#include <math.h>

// MI455X / gfx1250: wave32, WMMA 16x16x32 f16 -> f32
typedef __attribute__((ext_vector_type(16))) _Float16 v16h;
typedef __attribute__((ext_vector_type(8)))  _Float16 v8h;
typedef __attribute__((ext_vector_type(8)))  float    v8f;

#define BB 4
#define SS 2048
#define DD 128
#define HALFD 64
#define MIPE_THRESHOLD 1.5707963f
#define EPSV 1e-6f
#define KEYS_PER_WAVE 256           // key-chunk per wave in the fused kernel
#define NCHUNK (SS / KEYS_PER_WAVE) // 8

// ---------------------------------------------------------------------------
// Kernel 1: unit-normalize rows; apply MIPE rotation to q,k (f16 row-major);
// store v_unit transposed as f16 [b][d][s] so the aggregate GEMM's B-fragment
// (K = key contiguous per lane) loads are contiguous 32B.
// grid = (B*S, 3), block = 128
// ---------------------------------------------------------------------------
__global__ void su_preproc(const float* __restrict__ q,
                           const float* __restrict__ k,
                           const float* __restrict__ v,
                           const float* __restrict__ s_w_p,
                           _Float16* __restrict__ qm,
                           _Float16* __restrict__ km,
                           _Float16* __restrict__ vt) {
    const int row = blockIdx.x;
    const int t   = blockIdx.y;            // 0:q 1:k 2:v
    const int d   = threadIdx.x;           // 0..127
    const int b   = row / SS;
    const int s   = row % SS;
    const float* src = (t == 0) ? q : (t == 1) ? k : v;

    float x  = src[(size_t)row * DD + d];
    float ss = x * x;
    #pragma unroll
    for (int m = 16; m; m >>= 1) ss += __shfl_xor(ss, m, 32);
    __shared__ float red[4];
    if ((threadIdx.x & 31) == 0) red[threadIdx.x >> 5] = ss;
    __syncthreads();
    const float tot = red[0] + red[1] + red[2] + red[3];
    const float inv = 1.0f / (sqrtf(tot) + EPSV);

    if (t == 2) {
        vt[(size_t)b * DD * SS + (size_t)d * SS + s] = (_Float16)(x * inv);
        return;
    }
    const float window = expf(s_w_p[0]);
    const float theta  = fminf((float)s / window, MIPE_THRESHOLD);
    const float c  = cosf(theta);
    const float sn = sinf(theta);
    const int   pd = (d < HALFD) ? d + HALFD : d - HALFD;
    const float xu = x * inv;
    const float xp = src[(size_t)row * DD + pd] * inv;
    const float o  = (d < HALFD) ? (xu * c - xp * sn) : (xp * sn + xu * c);
    _Float16* dst = (t == 0) ? qm : km;
    dst[(size_t)row * DD + d] = (_Float16)o;
}

// ---------------------------------------------------------------------------
// Kernel 2: zero the partial-aggregate buffer (atomic accumulation target).
// ---------------------------------------------------------------------------
__global__ void su_zero(float* __restrict__ p, int n) {
    int i = blockIdx.x * 256 + threadIdx.x;
    if (i < n) p[i] = 0.0f;
}

// ---------------------------------------------------------------------------
// Kernel 3 (fused, phase-separated): one wave per (b, 16-q tile, 256-key
// chunk); block = 32 (EXEC all ones as WMMA requires).
// Phase 1: sim WMMA + elementwise -> NT f32 DAR store + f16 LDS tile
//          (no accumulators live; Aq fragments stay in registers).
// Phase 2: after one barrier, Aq is dead; 64-VGPR accumulator set, A-fragment
//          from LDS (two aligned b128 DS loads), 8 independent WMMA chains.
// ---------------------------------------------------------------------------
__global__ void su_fused(const _Float16* __restrict__ qm,
                         const _Float16* __restrict__ km,
                         const _Float16* __restrict__ vt,
                         const float* __restrict__ s_w_p,
                         const float* __restrict__ s_r_p,
                         float* __restrict__ dar,
                         float* __restrict__ agg) {
    const int nQT   = SS / 16;                 // 128
    const int id    = blockIdx.x;
    const int b     = id / (nQT * NCHUNK);
    const int rem   = id % (nQT * NCHUNK);
    const int qt    = rem / NCHUNK;
    const int chunk = rem % NCHUNK;
    const int l  = threadIdx.x;
    const int lg = l >> 4;                     // K-group of this lane
    const int ln = l & 15;

    // DAR f16 staging for the whole 256-key chunk: [m 0..15][key 0..255] (8KB)
    __shared__ _Float16 tile[16 * KEYS_PER_WAVE];

    const float window = expf(s_w_p[0]);
    const float r      = 1.0f / (1.0f + expf(-s_r_p[0]));
    const float invomr = 1.0f / (1.0f - r);

    // =============== Phase 1: similarity + relevance ===============
    {
        // A fragments: 16 q-rows x K=128 (4 x v16h)
        const _Float16* qrow = qm + ((size_t)b * SS + qt * 16 + ln) * DD;
        v16h Aq[4];
        #pragma unroll
        for (int kk = 0; kk < 4; ++kk) {
            v8h lo = *(const v8h*)(qrow + kk * 32 + lg * 16);
            v8h hi = *(const v8h*)(qrow + kk * 32 + lg * 16 + 8);
            #pragma unroll
            for (int i = 0; i < 8; ++i) { Aq[kk][i] = lo[i]; Aq[kk][8 + i] = hi[i]; }
        }

        // DAR row base for this (qtile, chunk): offset = (rI+8*lg)*SS + keyLocal
        float* drow = dar + ((size_t)b * SS + qt * 16) * SS + chunk * KEYS_PER_WAVE;

        for (int jt = 0; jt < KEYS_PER_WAVE / 16; ++jt) {
            const int kbase = chunk * KEYS_PER_WAVE + jt * 16;
            const _Float16* krow = km + ((size_t)b * SS + kbase + ln) * DD;
            v8f c = {};
            #pragma unroll
            for (int kk = 0; kk < 4; ++kk) {
                v16h Bv = *(const v16h*)(krow + kk * 32 + lg * 16);
                c = __builtin_amdgcn_wmma_f32_16x16x32_f16(
                        false, Aq[kk], false, Bv, (short)0, c, false, false);
            }
            #pragma unroll
            for (int rI = 0; rI < 8; ++rI) {
                const int m   = rI + 8 * lg;                 // row in tile
                const int qi  = qt * 16 + m;
                const int kj  = kbase + ln;
                float sim = fminf(fmaxf(c[rI], -1.0f), 1.0f);
                float rel = fmaxf(sim - r, 0.0f) * invomr;
                rel *= rel;
                const float dist = fabsf((float)(qi - kj));
                const float mask = 1.0f / (1.0f + expf(dist - window));
                const float dv   = rel * mask;
                __builtin_nontemporal_store(
                    dv, drow + (size_t)m * SS + jt * 16 + ln);
                tile[m * KEYS_PER_WAVE + jt * 16 + ln] = (_Float16)dv;
            }
        }
    }
    __syncthreads();   // Aq dies here; tile becomes readable

    // =============== Phase 2: aggregate over this key chunk ===============
    v8f acc[8];
    #pragma unroll
    for (int nt = 0; nt < 8; ++nt) acc[nt] = (v8f){};

    const _Float16* vbase = vt + (size_t)b * DD * SS + chunk * KEYS_PER_WAVE;

    for (int kt = 0; kt < KEYS_PER_WAVE / 32; ++kt) {
        // A fragment: rows=q (m=ln), K = 32 local keys, contiguous per lane
        v8h alo = *(const v8h*)(tile + ln * KEYS_PER_WAVE + kt * 32 + lg * 16);
        v8h ahi = *(const v8h*)(tile + ln * KEYS_PER_WAVE + kt * 32 + lg * 16 + 8);
        v16h Ad;
        #pragma unroll
        for (int i = 0; i < 8; ++i) { Ad[i] = alo[i]; Ad[8 + i] = ahi[i]; }

        // 8 independent accumulation chains over the 16-wide d-tiles
        #pragma unroll
        for (int nt = 0; nt < 8; ++nt) {
            const _Float16* vcol = vbase + ((size_t)(nt * 16 + ln)) * SS;
            v16h Bv = *(const v16h*)(vcol + kt * 32 + lg * 16);
            acc[nt] = __builtin_amdgcn_wmma_f32_16x16x32_f16(
                          false, Ad, false, Bv, (short)0, acc[nt], false, false);
        }
    }

    // combine partials across the 8 key-chunk waves
    #pragma unroll
    for (int nt = 0; nt < 8; ++nt) {
        #pragma unroll
        for (int rI = 0; rI < 8; ++rI) {
            const int qi = qt * 16 + rI + 8 * lg;
            const int dj = nt * 16 + ln;
            atomicAdd(&agg[((size_t)b * SS + qi) * DD + dj], acc[nt][rI]);
        }
    }
}

// ---------------------------------------------------------------------------
// Kernel 4: row-wise TanhNorm. grid = B*S, block = 128.
// ---------------------------------------------------------------------------
__global__ void su_tanh_norm(const float* __restrict__ agg,
                             float* __restrict__ out) {
    const int row = blockIdx.x;
    const int d   = threadIdx.x;
    const float a = agg[(size_t)row * DD + d];
    float ss = a * a;
    #pragma unroll
    for (int m = 16; m; m >>= 1) ss += __shfl_xor(ss, m, 32);
    __shared__ float red[4];
    if ((threadIdx.x & 31) == 0) red[threadIdx.x >> 5] = ss;
    __syncthreads();
    const float tot = red[0] + red[1] + red[2] + red[3];
    const float rms = sqrtf(tot / (float)DD + EPSV);
    out[(size_t)row * DD + d] = tanhf(a / rms);
}

// ---------------------------------------------------------------------------
extern "C" void kernel_launch(void* const* d_in, const int* in_sizes, int n_in,
                              void* d_out, int out_size, void* d_ws, size_t ws_size,
                              hipStream_t stream) {
    const float* q   = (const float*)d_in[0];
    const float* k   = (const float*)d_in[1];
    const float* v   = (const float*)d_in[2];
    const float* s_w = (const float*)d_in[3];
    const float* s_r = (const float*)d_in[4];

    float* out = (float*)d_out;                       // [B,S,D]
    float* dar = out + (size_t)BB * SS * DD;          // [B,S,S]

    // workspace: qm(2MB) km(2MB) vt(2MB) agg(4MB) = 10MB
    _Float16* qm  = (_Float16*)d_ws;
    _Float16* km  = qm + (size_t)BB * SS * DD;
    _Float16* vt  = km + (size_t)BB * SS * DD;
    float*    agg = (float*)(vt + (size_t)BB * SS * DD);
    const int aggN = BB * SS * DD;

    su_preproc<<<dim3(BB * SS, 3), 128, 0, stream>>>(q, k, v, s_w, qm, km, vt);

    su_zero<<<(aggN + 255) / 256, 256, 0, stream>>>(agg, aggN);

    su_fused<<<BB * (SS / 16) * NCHUNK, 32, 0, stream>>>(
        qm, km, vt, s_w, s_r, dar, agg);

    su_tanh_norm<<<BB * SS, 128, 0, stream>>>(agg, out);
}
